// ChebConv_Layer_54185307406514
// MI455X (gfx1250) — matrix operand order repaired and checked by reading, not compiled, not run on hardware
//
#include <hip/hip_runtime.h>

// Problem constants (match reference)
#define BATCH 8
#define NNODES 10000
#define NEDGES 320000
#define DIN 64
#define DOUT 64
#define KCAT 192           // 3 * DIN concatenated features
#define NTILES (NNODES / 16)   // 625

typedef __attribute__((ext_vector_type(16))) __bf16 v16bf;
typedef __attribute__((ext_vector_type(8)))  float  v8f;

// ---------------------------------------------------------------- zero
__global__ void zero_f32_kernel(float* __restrict__ p, long n) {
    long i = (long)blockIdx.x * blockDim.x + threadIdx.x;
    long stride = (long)gridDim.x * blockDim.x;
    for (; i < n; i += stride) p[i] = 0.0f;
}

// ---------------------------------------------------------------- degree
__global__ void deg_kernel(const int* __restrict__ row,
                           const float* __restrict__ attr,
                           float* __restrict__ deg) {
    int i = blockIdx.x * blockDim.x + threadIdx.x;
    if (i < NEDGES) atomicAdd(&deg[row[i]], attr[i]);
}

// ---------------------------------------------------------------- edge norm
__global__ void norm_kernel(const int* __restrict__ row,
                            const int* __restrict__ col,
                            const float* __restrict__ attr,
                            const float* __restrict__ deg,
                            float* __restrict__ nrm) {
    int i = blockIdx.x * blockDim.x + threadIdx.x;
    if (i >= NEDGES) return;
    float dr = deg[row[i]];
    float dc = deg[col[i]];
    float rr = dr > 0.0f ? rsqrtf(dr) : 0.0f;
    float rc = dc > 0.0f ? rsqrtf(dc) : 0.0f;
    nrm[i] = -(rr * attr[i] * rc);
}

// ------------------------------------------------- fused weight prep (bf16 split)
// Effective weights for folded GEMM over K=192:
//   seg0 (k<64):  W0 - W2   seg1: W1   seg2: 2*W2
// Stored column-major (n-major) so each WMMA B fragment is 16 contiguous bf16.
__global__ void wprep_kernel(const float* __restrict__ W,
                             __bf16* __restrict__ wt_hi,
                             __bf16* __restrict__ wt_lo) {
    int i = blockIdx.x * blockDim.x + threadIdx.x;
    if (i >= KCAT * DOUT) return;
    int k   = i >> 6;       // 0..191
    int n   = i & 63;       // 0..63
    int seg = k >> 6;       // 0,1,2
    int kk  = k & 63;
    float w0 = W[0 * DIN * DOUT + kk * DOUT + n];
    float w1 = W[1 * DIN * DOUT + kk * DOUT + n];
    float w2 = W[2 * DIN * DOUT + kk * DOUT + n];
    float val = (seg == 0) ? (w0 - w2) : ((seg == 1) ? w1 : 2.0f * w2);
    __bf16 h = (__bf16)val;
    float rem = val - (float)h;
    wt_hi[(size_t)n * KCAT + k] = h;
    wt_lo[(size_t)n * KCAT + k] = (__bf16)rem;
}

// ---------------------------------------------------------------- propagation
// One wave32 per (edge, batch); lane handles 2 consecutive features.
// dst[b, col[e], :] += src[b, row[e], :] * norm[e]
__global__ __launch_bounds__(256) void prop_kernel(
        const float* __restrict__ src, float* __restrict__ dst,
        const int* __restrict__ row, const int* __restrict__ col,
        const float* __restrict__ nrm) {
    long task = (((long)blockIdx.x * 256 + threadIdx.x) >> 5); // wave id
    if (task >= (long)NEDGES * BATCH) return;
    int lane = threadIdx.x & 31;
    int e = (int)(task >> 3);   // BATCH == 8
    int b = (int)(task & 7);
    int r = row[e];
    int c = col[e];
    float w = nrm[e];
    const float* s = src + (((size_t)b * NNODES + r) << 6) + lane * 2;
    float*       d = dst + (((size_t)b * NNODES + c) << 6) + lane * 2;
    float2 v = *(const float2*)s;
    atomicAdd(d + 0, v.x * w);
    atomicAdd(d + 1, v.y * w);
}

// ---------------------------------------------------------------- WMMA GEMM
// out[b, node0+0..15, 0..63] = [x | tx1 | tmp](16 x 192) @ Wcat(192 x 64) + bias
// 4 waves per block; each wave owns one 16-column block (one 16x16 D tile).
// bf16x3 split: C += Ah*Bh + Ah*Bl + Al*Bh  (f32 accumulate)
__global__ __launch_bounds__(128) void cheb_gemm_kernel(
        const float* __restrict__ x, const float* __restrict__ tx1,
        const float* __restrict__ tmp,
        const __bf16* __restrict__ wt_hi, const __bf16* __restrict__ wt_lo,
        const float* __restrict__ bias, float* __restrict__ out) {
    int tile  = blockIdx.x;           // 0 .. BATCH*NTILES-1
    int b     = tile / NTILES;
    int node0 = (tile % NTILES) * 16;
    int wave  = threadIdx.x >> 5;     // 0..3 -> column block
    int lane  = threadIdx.x & 31;
    int half  = lane >> 4;            // 0/1
    int mn    = lane & 15;            // A-row index / B,D-column index
    int col0  = wave * 16;

    const float* src0 = x   + ((size_t)b * NNODES + node0) * 64;
    const float* src1 = tx1 + ((size_t)b * NNODES + node0) * 64;
    const float* src2 = tmp + ((size_t)b * NNODES + node0) * 64;

    v8f c = {};
    #pragma unroll
    for (int ks = 0; ks < 6; ++ks) {
        const float* base = (ks < 2) ? src0 : ((ks < 4) ? src1 : src2);
        const float* srow = base + (size_t)mn * 64 + (ks & 1) * 32;

        // A fragment (16x32 bf16): elems 0..7 -> K = 16*half + j,
        //                          elems 8..15 -> K = 16*half + 8 + j
        v16bf ah, al;
        #pragma unroll
        for (int j = 0; j < 8; ++j) {
            float v0 = srow[half * 16 + j];
            float v1 = srow[half * 16 + 8 + j];
            __bf16 h0 = (__bf16)v0;
            __bf16 h1 = (__bf16)v1;
            ah[j]     = h0;  al[j]     = (__bf16)(v0 - (float)h0);
            ah[8 + j] = h1;  al[8 + j] = (__bf16)(v1 - (float)h1);
        }

        // B fragment (32x16 bf16): lane column = col0+mn,
        // 16 contiguous K values starting at ks*32 + 16*half (column-major store)
        size_t boff = (size_t)(col0 + mn) * KCAT + ks * 32 + half * 16;
        v16bf vbh = *(const v16bf*)(wt_hi + boff);
        v16bf vbl = *(const v16bf*)(wt_lo + boff);

        c = __builtin_amdgcn_wmma_f32_16x16x32_bf16(false, ah, false, vbh,
                                                    (short)0, c, false, false);
        c = __builtin_amdgcn_wmma_f32_16x16x32_bf16(false, ah, false, vbl,
                                                    (short)0, c, false, false);
        c = __builtin_amdgcn_wmma_f32_16x16x32_bf16(false, al, false, vbh,
                                                    (short)0, c, false, false);
    }

    float bv = bias[col0 + mn];
    #pragma unroll
    for (int v = 0; v < 8; ++v) {
        int m = v + 8 * half;
        out[(((size_t)b * NNODES + node0 + m) << 6) + col0 + mn] = c[v] + bv;
    }
}

// ---------------------------------------------------------------- launch
extern "C" void kernel_launch(void* const* d_in, const int* in_sizes, int n_in,
                              void* d_out, int out_size, void* d_ws, size_t ws_size,
                              hipStream_t stream) {
    const float* x    = (const float*)d_in[0];           // [B, N, 64]
    const int*   ei   = (const int*)d_in[1];             // [2, E]
    const float* attr = (const float*)d_in[2];           // [E]
    const float* W    = (const float*)d_in[3];           // [3, 64, 64]
    const float* bias = (const float*)d_in[4];           // [64]
    float* out = (float*)d_out;
    const int* row = ei;
    const int* col = ei + NEDGES;

    // Workspace layout (all 256B aligned)
    char* ws = (char*)d_ws;
    const size_t SZ_TX  = (size_t)BATCH * NNODES * 64 * sizeof(float); // 20,480,000
    float*  tx1   = (float*)(ws);
    float*  tmp   = (float*)(ws + SZ_TX);
    float*  deg   = (float*)(ws + 2 * SZ_TX);                          // N floats
    float*  nrm   = (float*)(ws + 2 * SZ_TX + 40192);                  // E floats
    __bf16* wt_hi = (__bf16*)(ws + 2 * SZ_TX + 40192 + 1280000);
    __bf16* wt_lo = (__bf16*)(ws + 2 * SZ_TX + 40192 + 1280000 + 24576);

    // 1. Zero accumulators (tx1, tmp, deg are contiguous)
    long nzero = 2L * BATCH * NNODES * 64 + NNODES;
    zero_f32_kernel<<<8192, 256, 0, stream>>>(tx1, nzero);

    // 2. Weighted out-degree
    deg_kernel<<<(NEDGES + 255) / 256, 256, 0, stream>>>(row, attr, deg);

    // 3. Edge normalization: -d_r^-1/2 * w * d_c^-1/2
    norm_kernel<<<(NEDGES + 255) / 256, 256, 0, stream>>>(row, col, attr, deg, nrm);

    // 4. Fused weights -> bf16 hi/lo split, column-major
    wprep_kernel<<<(KCAT * DOUT + 255) / 256, 256, 0, stream>>>(W, wt_hi, wt_lo);

    // 5. Tx1 = prop(x)
    {
        long waves = (long)NEDGES * BATCH;           // 2,560,000
        int blocks = (int)((waves * 32 + 255) / 256); // 320,000
        prop_kernel<<<blocks, 256, 0, stream>>>(x, tx1, row, col, nrm);
        // 6. tmp = prop(Tx1)
        prop_kernel<<<blocks, 256, 0, stream>>>(tx1, tmp, row, col, nrm);
    }

    // 7. out = x@(W0-W2) + Tx1@W1 + tmp@(2W2) + bias   (WMMA bf16x3)
    cheb_gemm_kernel<<<BATCH * NTILES, 128, 0, stream>>>(x, tx1, tmp,
                                                         wt_hi, wt_lo, bias, out);
}
